// MessagePassingCouplingLayer_26998164422861
// MI455X (gfx1250) — matrix operand order repaired and checked by reading, not compiled, not run on hardware
//
#include <hip/hip_runtime.h>

#define B_DIM 16
#define N_DIM 4096
#define E_DIM 524288
#define H_DIM 128
#define NODES (B_DIM * N_DIM)

// Padded LDS row strides (in halves): (stride/2) mod 64 == odd multiple of 4
// so the 16 fragment-loading lanes spread across all 64 LDS banks.
#define LDA256 264
#define LDA128 136
#define LDA160 168
#define LDA96  104
#define LDA64  72

typedef __attribute__((ext_vector_type(16))) _Float16 v16h;
typedef __attribute__((ext_vector_type(8)))  _Float16 v8h;
typedef __attribute__((ext_vector_type(8)))  float    v8f;

__device__ __forceinline__ v8f v8f_zero() {
  v8f v;
#pragma unroll
  for (int i = 0; i < 8; ++i) v[i] = 0.0f;
  return v;
}

__device__ __forceinline__ v8f wmma_f16(v16h a, v16h b, v8f c) {
  return __builtin_amdgcn_wmma_f32_16x16x32_f16(false, a, false, b, (short)0, c,
                                                false, false);
}

// Tile GEMM: 256 threads = 8 waves. Wave w computes rows [16w,16w+16) of the
// 128-row tile times NT*16 output columns, accumulating over KC chunks of K=32.
// A: LDS row-major [128][lda] f16 (this code's contiguous k order: lanes 0-15 K{0..15},
// lanes 16-31 K{16..31} -> two ds_load_b128 per fragment).
// B: [col][bstride] f16 (LDS or L2-resident global) -> 32B contiguous per frag.
template <int KC, int NT>
__device__ __forceinline__ void gemm_tile(const _Float16* __restrict__ ldsA,
                                          int lda,
                                          const _Float16* __restrict__ B,
                                          int bstride, v8f* acc) {
  const int lane = threadIdx.x & 31;
  const int wave = threadIdx.x >> 5;
  const int arow = wave * 16 + (lane & 15);
  const int aoff = (lane < 16) ? 0 : 16;
  const int boff = (lane < 16) ? 0 : 16;
#pragma unroll
  for (int kk = 0; kk < KC; ++kk) {
    union { v16h v; v8h h[2]; } ua;
    const _Float16* ap = ldsA + arow * lda + kk * 32 + aoff;
    ua.h[0] = *(const v8h*)ap;
    ua.h[1] = *(const v8h*)(ap + 8);
#pragma unroll
    for (int nt = 0; nt < NT; ++nt) {
      const _Float16* bp =
          B + (size_t)(nt * 16 + (lane & 15)) * bstride + kk * 32 + boff;
      v16h bv = *(const v16h*)bp;
      acc[nt] = wmma_f16(ua.v, bv, acc[nt]);
    }
  }
}

// ---- weight pack: out[n*kpad + k] = f16(W[k*nc + n]), zero-pad k>=K ----
__global__ __launch_bounds__(256) void mpcl_pack_wT(const float* __restrict__ W,
                                                    _Float16* __restrict__ out,
                                                    int K, int nc, int kpad) {
  int i = blockIdx.x * 256 + threadIdx.x;
  if (i >= nc * kpad) return;
  int n = i / kpad, k = i - n * kpad;
  float v = (k < K) ? W[(size_t)k * nc + n] : 0.0f;
  out[i] = (_Float16)v;
}

// ---- node init: hf = relu([embed[type], xyz] @ W_in + b) * valid ----
__global__ __launch_bounds__(256) void mpcl_node_init(
    const float* __restrict__ embed, const int* __restrict__ types,
    const float* __restrict__ coords, const unsigned char* __restrict__ masked,
    const _Float16* __restrict__ wInT, const float* __restrict__ b_in,
    _Float16* __restrict__ hf) {
  __shared__ _Float16 sA[128 * LDA96];
  const int tid = threadIdx.x;
  const int base = blockIdx.x * 128;
  for (int i = tid; i < 128 * 96; i += 256) {
    int r = i / 96, c = i - r * 96;
    int node = base + r;
    float v = 0.0f;
    if (c < 64)      v = embed[(size_t)types[node] * 64 + c];
    else if (c < 67) v = coords[(size_t)node * 3 + (c - 64)];
    sA[r * LDA96 + c] = (_Float16)v;
  }
  __syncthreads();
  v8f acc[8];
#pragma unroll
  for (int nt = 0; nt < 8; ++nt) acc[nt] = v8f_zero();
  gemm_tile<3, 8>(sA, LDA96, wInT, 96, acc);
  const int lane = tid & 31, wave = tid >> 5;
#pragma unroll
  for (int nt = 0; nt < 8; ++nt) {
    int col = nt * 16 + (lane & 15);
    float bv = b_in[col];
#pragma unroll
    for (int g = 0; g < 8; ++g) {
      int r = wave * 16 + ((lane < 16) ? g : 8 + g);
      int node = base + r;
      float v = fmaxf(acc[nt][g] + bv, 0.0f);
      if (masked[node]) v = 0.0f;
      hf[(size_t)node * 128 + col] = (_Float16)v;
    }
  }
}

// ---- edge messages + fused scatter segment_sum (weights staged in LDS) ----
__global__ __launch_bounds__(256) void mpcl_edge_msg(
    const _Float16* __restrict__ hf, const int* __restrict__ adj,
    const int* __restrict__ ebi, const _Float16* __restrict__ wT,
    const float* __restrict__ bias, float* __restrict__ agg) {
  __shared__ _Float16 sA[128 * LDA256];  // [edge][hf[src]|hf[dst]]
  __shared__ _Float16 sB[128 * LDA256];  // [col][K=256]
  __shared__ int sSrc[128];
  __shared__ int sDst[128];
  const int tid = threadIdx.x;
  const int e0 = blockIdx.x * 128;
  if (tid == 0 && (int)blockIdx.x + 1 < (int)gridDim.x)
    __builtin_prefetch(adj + 2 * (e0 + 128), 0, 1);  // global_prefetch_b8
  if (tid < 128) {
    int e = e0 + tid;
    int bb = ebi[e];
    sSrc[tid] = bb * N_DIM + adj[2 * e];
    sDst[tid] = bb * N_DIM + adj[2 * e + 1];
  }
  for (int i = tid; i < 128 * 32; i += 256) {  // weights -> LDS (64KB, once)
    int col = i >> 5, j = i & 31;
    ((uint4*)(sB + col * LDA256))[j] = ((const uint4*)(wT + (size_t)col * 256))[j];
  }
  __syncthreads();
  for (int i = tid; i < 128 * 32; i += 256) {
    int r = i >> 5, j = i & 31;
    const uint4* src = (j < 16)
        ? ((const uint4*)(hf + (size_t)sSrc[r] * 128) + j)
        : ((const uint4*)(hf + (size_t)sDst[r] * 128) + (j - 16));
    ((uint4*)(sA + r * LDA256))[j] = *src;
  }
  __syncthreads();
  v8f acc[8];
#pragma unroll
  for (int nt = 0; nt < 8; ++nt) acc[nt] = v8f_zero();
  gemm_tile<8, 8>(sA, LDA256, sB, LDA256, acc);
  const int lane = tid & 31, wave = tid >> 5;
#pragma unroll
  for (int nt = 0; nt < 8; ++nt) {
    int col = nt * 16 + (lane & 15);
    float bv = bias[col];
#pragma unroll
    for (int g = 0; g < 8; ++g) {
      int r = wave * 16 + ((lane < 16) ? g : 8 + g);
      float v = fmaxf(acc[nt][g] + bv, 0.0f);
      unsafeAtomicAdd(&agg[(size_t)sDst[r] * 128 + col], v);
    }
  }
}

// ---- node update: hf' = relu([hf, agg] @ W_upd + b) (weights in LDS) ----
__global__ __launch_bounds__(256) void mpcl_node_upd(
    const _Float16* __restrict__ hfin, const float* __restrict__ agg,
    const _Float16* __restrict__ wT, const float* __restrict__ bias,
    _Float16* __restrict__ hfout) {
  __shared__ _Float16 sA[128 * LDA256];
  __shared__ _Float16 sB[128 * LDA256];
  const int tid = threadIdx.x;
  const int base = blockIdx.x * 128;
  for (int i = tid; i < 128 * 32; i += 256) {
    int col = i >> 5, j = i & 31;
    ((uint4*)(sB + col * LDA256))[j] = ((const uint4*)(wT + (size_t)col * 256))[j];
  }
  for (int i = tid; i < 128 * 16; i += 256) {
    int r = i >> 4, j = i & 15;
    ((uint4*)(sA + r * LDA256))[j] =
        ((const uint4*)(hfin + (size_t)(base + r) * 128))[j];
  }
  for (int i = tid; i < 128 * 128; i += 256) {
    int r = i >> 7, c = i & 127;
    sA[r * LDA256 + 128 + c] = (_Float16)agg[(size_t)(base + r) * 128 + c];
  }
  __syncthreads();
  v8f acc[8];
#pragma unroll
  for (int nt = 0; nt < 8; ++nt) acc[nt] = v8f_zero();
  gemm_tile<8, 8>(sA, LDA256, sB, LDA256, acc);
  const int lane = tid & 31, wave = tid >> 5;
#pragma unroll
  for (int nt = 0; nt < 8; ++nt) {
    int col = nt * 16 + (lane & 15);
    float bv = bias[col];
#pragma unroll
    for (int g = 0; g < 8; ++g) {
      int r = wave * 16 + ((lane < 16) ? g : 8 + g);
      float v = fmaxf(acc[nt][g] + bv, 0.0f);
      hfout[(size_t)(base + r) * 128 + col] = (_Float16)v;
    }
  }
}

// ---- fused out-proj + coupling MLPs + transform + log_det ----
__global__ __launch_bounds__(256) void mpcl_final(
    const _Float16* __restrict__ hf, const _Float16* __restrict__ wOutT,
    const float* __restrict__ b_out, const float* __restrict__ coords,
    const int* __restrict__ types, const unsigned char* __restrict__ masked,
    const _Float16* __restrict__ wS0T, const float* __restrict__ bs0,
    const _Float16* __restrict__ wS1T, const float* __restrict__ bs1,
    const _Float16* __restrict__ wS2T, const float* __restrict__ bs2,
    const float* __restrict__ Ws3, const float* __restrict__ bs3,
    const _Float16* __restrict__ wT0T, const float* __restrict__ bt0,
    const _Float16* __restrict__ wT1T, const float* __restrict__ bt1,
    const _Float16* __restrict__ wT2T, const float* __restrict__ bt2,
    const float* __restrict__ Wt3, const float* __restrict__ bt3,
    float* __restrict__ out, float* __restrict__ logdet) {
  __shared__ _Float16 sH[128 * LDA128];   // staged hf tile, later reused as Y1
  __shared__ _Float16 sX[128 * LDA160];   // [node_feats(128)|cond(3)|temps(2)|pad]
  __shared__ _Float16 sY0[128 * LDA128];
  __shared__ _Float16 sY2[128 * LDA64];
  __shared__ float    sR[2][128][3];
  __shared__ float    sRed[128];
  const int tid = threadIdx.x;
  const int base = blockIdx.x * 128;
  const int batch = base / N_DIM;  // tile never crosses a batch (4096 % 128 == 0)
  const int lane = tid & 31, wave = tid >> 5;

  for (int i = tid; i < 128 * 16; i += 256) {
    int r = i >> 4, j = i & 15;
    ((uint4*)(sH + r * LDA128))[j] =
        ((const uint4*)(hf + (size_t)(base + r) * 128))[j];
  }
  __syncthreads();
  {  // node_feats = hf @ W_out + b_out, masked -> sX[:,0:128]
    v8f acc[8];
#pragma unroll
    for (int nt = 0; nt < 8; ++nt) acc[nt] = v8f_zero();
    gemm_tile<4, 8>(sH, LDA128, wOutT, 128, acc);
#pragma unroll
    for (int nt = 0; nt < 8; ++nt) {
      int col = nt * 16 + (lane & 15);
      float bv = b_out[col];
#pragma unroll
      for (int g = 0; g < 8; ++g) {
        int r = wave * 16 + ((lane < 16) ? g : 8 + g);
        float v = acc[nt][g] + bv;
        if (masked[base + r]) v = 0.0f;
        sX[r * LDA160 + col] = (_Float16)v;
      }
    }
  }
  for (int i = tid; i < 128 * 32; i += 256) {  // cond + temps + pad
    int r = i >> 5, c = 128 + (i & 31);
    int node = base + r;
    float v = 0.0f;
    if (c < 131) {
      bool coup = (types[node] > 0) && !masked[node];
      v = coup ? 0.0f : coords[(size_t)node * 3 + (c - 128)];
    } else if (c == 131) v = 300.0f;
    else if (c == 132)   v = 600.0f;
    sX[r * LDA160 + c] = (_Float16)v;
  }
  __syncthreads();

  for (int mlp = 0; mlp < 2; ++mlp) {
    const _Float16* w0 = mlp ? wT0T : wS0T; const float* l0b = mlp ? bt0 : bs0;
    const _Float16* w1 = mlp ? wT1T : wS1T; const float* l1b = mlp ? bt1 : bs1;
    const _Float16* w2 = mlp ? wT2T : wS2T; const float* l2b = mlp ? bt2 : bs2;
    const float*    w3 = mlp ? Wt3  : Ws3;  const float* l3b = mlp ? bt3 : bs3;
    {  // L0: 160 -> 128
      v8f acc[8];
#pragma unroll
      for (int nt = 0; nt < 8; ++nt) acc[nt] = v8f_zero();
      gemm_tile<5, 8>(sX, LDA160, w0, 160, acc);
#pragma unroll
      for (int nt = 0; nt < 8; ++nt) {
        int col = nt * 16 + (lane & 15);
        float bv = l0b[col];
#pragma unroll
        for (int g = 0; g < 8; ++g) {
          int r = wave * 16 + ((lane < 16) ? g : 8 + g);
          sY0[r * LDA128 + col] = (_Float16)fmaxf(acc[nt][g] + bv, 0.0f);
        }
      }
    }
    __syncthreads();
    {  // L1: 128 -> 128 (write into sH, free after out-proj)
      v8f acc[8];
#pragma unroll
      for (int nt = 0; nt < 8; ++nt) acc[nt] = v8f_zero();
      gemm_tile<4, 8>(sY0, LDA128, w1, 128, acc);
#pragma unroll
      for (int nt = 0; nt < 8; ++nt) {
        int col = nt * 16 + (lane & 15);
        float bv = l1b[col];
#pragma unroll
        for (int g = 0; g < 8; ++g) {
          int r = wave * 16 + ((lane < 16) ? g : 8 + g);
          sH[r * LDA128 + col] = (_Float16)fmaxf(acc[nt][g] + bv, 0.0f);
        }
      }
    }
    __syncthreads();
    {  // L2: 128 -> 64
      v8f acc[4];
#pragma unroll
      for (int nt = 0; nt < 4; ++nt) acc[nt] = v8f_zero();
      gemm_tile<4, 4>(sH, LDA128, w2, 128, acc);
#pragma unroll
      for (int nt = 0; nt < 4; ++nt) {
        int col = nt * 16 + (lane & 15);
        float bv = l2b[col];
#pragma unroll
        for (int g = 0; g < 8; ++g) {
          int r = wave * 16 + ((lane < 16) ? g : 8 + g);
          sY2[r * LDA64 + col] = (_Float16)fmaxf(acc[nt][g] + bv, 0.0f);
        }
      }
    }
    __syncthreads();
    if (tid < 128) {  // L3: 64 -> 3
      float o0 = l3b[0], o1 = l3b[1], o2 = l3b[2];
#pragma unroll 8
      for (int k = 0; k < 64; ++k) {
        float y = (float)sY2[tid * LDA64 + k];
        o0 += y * w3[k * 3 + 0];
        o1 += y * w3[k * 3 + 1];
        o2 += y * w3[k * 3 + 2];
      }
      sR[mlp][tid][0] = o0; sR[mlp][tid][1] = o1; sR[mlp][tid][2] = o2;
    }
    __syncthreads();
  }

  float ld = 0.0f;
  if (tid < 128) {
    int node = base + tid;
    bool valid = !masked[node];
    bool coup = (types[node] > 0) && valid;
#pragma unroll
    for (int j = 0; j < 3; ++j) {
      float x = coords[(size_t)node * 3 + j];
      float ls = coup ? tanhf(sR[0][tid][j]) * 0.5f : 0.0f;
      float sh = coup ? sR[1][tid][j] : 0.0f;
      out[(size_t)node * 3 + j] = coup ? __expf(ls) * x + sh : x;
      ld += ls;
    }
    sRed[tid] = ld;
  }
  __syncthreads();
  if (tid == 0) {
    float s = 0.0f;
    for (int r = 0; r < 128; ++r) s += sRed[r];
    unsafeAtomicAdd(&logdet[batch], s);
  }
}

extern "C" void kernel_launch(void* const* d_in, const int* in_sizes, int n_in,
                              void* d_out, int out_size, void* d_ws,
                              size_t ws_size, hipStream_t stream) {
  const float* coords = (const float*)d_in[0];
  const int* types    = (const int*)d_in[1];
  const int* adj      = (const int*)d_in[2];
  const int* ebi      = (const int*)d_in[3];
  const unsigned char* masked = (const unsigned char*)d_in[4];
  const float* embed  = (const float*)d_in[5];
  const float* W_in = (const float*)d_in[6];   const float* b_in = (const float*)d_in[7];
  const float* Wm0  = (const float*)d_in[8];   const float* bm0  = (const float*)d_in[9];
  const float* Wu0  = (const float*)d_in[10];  const float* bu0  = (const float*)d_in[11];
  const float* Wm1  = (const float*)d_in[12];  const float* bm1  = (const float*)d_in[13];
  const float* Wu1  = (const float*)d_in[14];  const float* bu1  = (const float*)d_in[15];
  const float* Wout = (const float*)d_in[16];  const float* bout = (const float*)d_in[17];
  const float* Ws0 = (const float*)d_in[18];   const float* bs0 = (const float*)d_in[19];
  const float* Ws1 = (const float*)d_in[20];   const float* bs1 = (const float*)d_in[21];
  const float* Ws2 = (const float*)d_in[22];   const float* bs2 = (const float*)d_in[23];
  const float* Ws3 = (const float*)d_in[24];   const float* bs3 = (const float*)d_in[25];
  const float* Wt0 = (const float*)d_in[26];   const float* bt0 = (const float*)d_in[27];
  const float* Wt1 = (const float*)d_in[28];   const float* bt1 = (const float*)d_in[29];
  const float* Wt2 = (const float*)d_in[30];   const float* bt2 = (const float*)d_in[31];
  const float* Wt3 = (const float*)d_in[32];   const float* bt3 = (const float*)d_in[33];
  float* out = (float*)d_out;
  float* logdet = out + (size_t)NODES * 3;

  // ---- carve workspace ----
  char* ws = (char*)d_ws;
  auto carve = [&](size_t bytes) {
    char* p = ws;
    ws += (bytes + 255) & ~(size_t)255;
    return p;
  };
  _Float16* wInT  = (_Float16*)carve(128 * 96 * 2);
  _Float16* wM0T  = (_Float16*)carve(128 * 256 * 2);
  _Float16* wU0T  = (_Float16*)carve(128 * 256 * 2);
  _Float16* wM1T  = (_Float16*)carve(128 * 256 * 2);
  _Float16* wU1T  = (_Float16*)carve(128 * 256 * 2);
  _Float16* wOutT = (_Float16*)carve(128 * 128 * 2);
  _Float16* wS0T  = (_Float16*)carve(128 * 160 * 2);
  _Float16* wS1T  = (_Float16*)carve(128 * 128 * 2);
  _Float16* wS2T  = (_Float16*)carve(64 * 128 * 2);
  _Float16* wT0T  = (_Float16*)carve(128 * 160 * 2);
  _Float16* wT1T  = (_Float16*)carve(128 * 128 * 2);
  _Float16* wT2T  = (_Float16*)carve(64 * 128 * 2);
  _Float16* hfA   = (_Float16*)carve((size_t)NODES * 128 * 2);
  _Float16* hfB   = (_Float16*)carve((size_t)NODES * 128 * 2);
  float*    agg   = (float*)carve((size_t)NODES * 128 * 4);

  auto packT = [&](const float* W, _Float16* o, int K, int nc, int kpad) {
    int total = nc * kpad;
    mpcl_pack_wT<<<(total + 255) / 256, 256, 0, stream>>>(W, o, K, nc, kpad);
  };
  packT(W_in, wInT, 67, 128, 96);
  packT(Wm0, wM0T, 256, 128, 256);
  packT(Wu0, wU0T, 256, 128, 256);
  packT(Wm1, wM1T, 256, 128, 256);
  packT(Wu1, wU1T, 256, 128, 256);
  packT(Wout, wOutT, 128, 128, 128);
  packT(Ws0, wS0T, 133, 128, 160);
  packT(Ws1, wS1T, 128, 128, 128);
  packT(Ws2, wS2T, 128, 64, 128);
  packT(Wt0, wT0T, 133, 128, 160);
  packT(Wt1, wT1T, 128, 128, 128);
  packT(Wt2, wT2T, 128, 64, 128);

  const int nodeTiles = NODES / 128;  // 512
  const int edgeTiles = E_DIM / 128;  // 4096

  mpcl_node_init<<<nodeTiles, 256, 0, stream>>>(embed, types, coords, masked,
                                                wInT, b_in, hfA);
  // layer 0
  hipMemsetAsync(agg, 0, (size_t)NODES * 128 * 4, stream);
  mpcl_edge_msg<<<edgeTiles, 256, 0, stream>>>(hfA, adj, ebi, wM0T, bm0, agg);
  mpcl_node_upd<<<nodeTiles, 256, 0, stream>>>(hfA, agg, wU0T, bu0, hfB);
  // layer 1
  hipMemsetAsync(agg, 0, (size_t)NODES * 128 * 4, stream);
  mpcl_edge_msg<<<edgeTiles, 256, 0, stream>>>(hfB, adj, ebi, wM1T, bm1, agg);
  mpcl_node_upd<<<nodeTiles, 256, 0, stream>>>(hfB, agg, wU1T, bu1, hfA);
  // output head
  hipMemsetAsync(logdet, 0, B_DIM * sizeof(float), stream);
  mpcl_final<<<nodeTiles, 256, 0, stream>>>(
      hfA, wOutT, bout, coords, types, masked, wS0T, bs0, wS1T, bs1, wS2T, bs2,
      Ws3, bs3, wT0T, bt0, wT1T, bt1, wT2T, bt2, Wt3, bt3, out, logdet);
}
